// AGLISTA_40553081209424
// MI455X (gfx1250) — matrix operand order repaired and checked by reading, not compiled, not run on hardware
//
#include <hip/hip_runtime.h>

#define MDIM 512
#define NDIM 2048
#define BDIM 2048
#define KITER 16

typedef __attribute__((ext_vector_type(16))) _Float16 v16h;
typedef __attribute__((ext_vector_type(8)))  _Float16 v8h;
typedef __attribute__((ext_vector_type(4)))  _Float16 v4h;
typedef __attribute__((ext_vector_type(8)))  float    v8f;
typedef __attribute__((ext_vector_type(4)))  float    v4f;

// ---------------- helpers ----------------

__device__ __forceinline__ v16h ld_frag(const _Float16* p, int gap) {
  // concat two 16B half-vectors into one 32B WMMA operand fragment
  return __builtin_shufflevector(*(const v8h*)p, *(const v8h*)(p + gap),
                                 0,1,2,3,4,5,6,7,8,9,10,11,12,13,14,15);
}

__device__ __forceinline__ v8f wmma_f16(v16h a, v16h b, v8f c) {
  return __builtin_amdgcn_wmma_f32_16x16x32_f16(
      /*neg_a=*/false, a, /*neg_b=*/false, b,
      /*c_mod=*/(short)0, c, /*reuse_a=*/false, /*reuse_b=*/false);
}

// epilogue GEMM1: Rh = f16(acc - y), 8 consecutive m per lane
__device__ __forceinline__ void store_r_tile(const v8f& acc, const float* yp,
                                             _Float16* rp) {
  v4f y0 = *(const v4f*)yp;
  v4f y1 = *(const v4f*)(yp + 4);
  v8h o;
#pragma unroll
  for (int j = 0; j < 4; ++j) {
    o[j]     = (_Float16)(acc[j]     - y0[j]);
    o[j + 4] = (_Float16)(acc[j + 4] - y1[j]);
  }
  *(v8h*)rp = o;
}

// epilogue GEMM2: xpre = x - gamma*acc, 8 consecutive n per lane
__device__ __forceinline__ void store_xpre_tile(const v8f& acc, float g,
                                                const float* xp, float* op) {
  v4f x0 = *(const v4f*)xp;
  v4f x1 = *(const v4f*)(xp + 4);
  v4f o0, o1;
#pragma unroll
  for (int j = 0; j < 4; ++j) {
    o0[j] = x0[j] - g * acc[j];
    o1[j] = x1[j] - g * acc[j + 4];
  }
  *(v4f*)op       = o0;
  *(v4f*)(op + 4) = o1;
}

// ---------------- setup kernels ----------------

__global__ __launch_bounds__(256) void zero_kernel(float* __restrict__ x) {
  int idx = blockIdx.x * 256 + threadIdx.x;        // over B*N/4
  v4f z = {0.f, 0.f, 0.f, 0.f};
  ((v4f*)x)[idx] = z;
}

__global__ __launch_bounds__(256) void prep_kernel(const float* __restrict__ phi,
                                                   const float* __restrict__ W,
                                                   _Float16* __restrict__ phiH,
                                                   _Float16* __restrict__ WT) {
  int idx = blockIdx.x * 256 + threadIdx.x;        // over M*N
  int m = idx / NDIM, n = idx % NDIM;
  phiH[idx] = (_Float16)phi[idx];
  WT[(size_t)n * MDIM + m] = (_Float16)W[idx];
}

// u = (1 + t*vu*exp(-v*|x|)) * x, f16, layout (B, N)
__global__ __launch_bounds__(256) void gain_kernel(
    const float* __restrict__ x, const float* __restrict__ theta,
    const float* __restrict__ theta_init, const float* __restrict__ v_arr,
    const float* __restrict__ vu, int it, _Float16* __restrict__ U) {
  int idx = blockIdx.x * 256 + threadIdx.x;        // over B*N/4
  float t  = (it == 0) ? theta_init[0] : theta[it];
  float tv = t * vu[it];
  float vv = v_arr[it];
  v4f xv = ((const v4f*)x)[idx];
  v4h o;
#pragma unroll
  for (int j = 0; j < 4; ++j) {
    float g = 1.0f + tv * __expf(-vv * fabsf(xv[j]));
    o[j] = (_Float16)(g * xv[j]);
  }
  ((v4h*)U)[idx] = o;
}

// ---------------- GEMM1: Rh(B,M) = f16( phi(M,N) @ u(N,B) - yT ) ----------------
// wave computes a 32m x 64b tile (2x4 WMMA tiles); K loop over N.

__global__ __launch_bounds__(128) void gemm1_kernel(
    const _Float16* __restrict__ phiH,  // M x N row-major
    const _Float16* __restrict__ U,     // B x N row-major (= u^T)
    const float*    __restrict__ y,     // B x M row-major (= yT^T)
    _Float16*       __restrict__ Rh) {  // B x M row-major
  const int lane = threadIdx.x & 31;
  const int wave = blockIdx.x * 4 + (threadIdx.x >> 5);
  const int tm = wave & 15;            // M/32 = 16 tiles
  const int tb = wave >> 4;            // B/64 = 32 tiles
  const int m0 = tm * 32, b0 = tb * 64;
  const int half = lane >> 4, l16 = lane & 15;

  v8f acc[2][4] = {};
  const _Float16* ap[2];
  const _Float16* bp[4];
  ap[0] = phiH + (size_t)(m0 + l16) * NDIM + half * 8;
  ap[1] = ap[0] + (size_t)16 * NDIM;
  bp[0] = U + (size_t)(b0 + l16) * NDIM + half * 8;
#pragma unroll
  for (int bi = 1; bi < 4; ++bi) bp[bi] = bp[bi - 1] + (size_t)16 * NDIM;

  for (int n0 = 0; n0 < NDIM; n0 += 32) {
    v16h a[2], bb[4];
#pragma unroll
    for (int mi = 0; mi < 2; ++mi) a[mi] = ld_frag(ap[mi] + n0, 16);
#pragma unroll
    for (int bi = 0; bi < 4; ++bi) bb[bi] = ld_frag(bp[bi] + n0, 16);
#pragma unroll
    for (int mi = 0; mi < 2; ++mi)
#pragma unroll
      for (int bi = 0; bi < 4; ++bi)
        acc[mi][bi] = wmma_f16(a[mi], bb[bi], acc[mi][bi]);
  }

#pragma unroll
  for (int mi = 0; mi < 2; ++mi)
#pragma unroll
    for (int bi = 0; bi < 4; ++bi) {
      int b = b0 + bi * 16 + l16;
      int m = m0 + mi * 16 + half * 8;
      store_r_tile(acc[mi][bi], y + (size_t)b * MDIM + m,
                   Rh + (size_t)b * MDIM + m);
    }
}

// ---------------- GEMM2: xpre(B,N) = x - gamma * ( W^T(N,M) @ R(M,B) ) ----------------
// wave computes a 32n x 64b tile (2x4 WMMA tiles); K loop over M.

__global__ __launch_bounds__(128) void gemm2_kernel(
    const _Float16* __restrict__ WT,    // N x M row-major
    const _Float16* __restrict__ Rh,    // B x M row-major
    const float*    __restrict__ x,     // B x N row-major
    const float*    __restrict__ gamma, int it,
    float*          __restrict__ Xpre) {// B x N row-major
  const int lane = threadIdx.x & 31;
  const int wave = blockIdx.x * 4 + (threadIdx.x >> 5);
  const int tn = wave & 63;            // N/32 = 64 tiles
  const int tb = wave >> 6;            // B/64 = 32 tiles
  const int n0 = tn * 32, b0 = tb * 64;
  const int half = lane >> 4, l16 = lane & 15;

  v8f acc[2][4] = {};
  const _Float16* ap[2];
  const _Float16* bp[4];
  ap[0] = WT + (size_t)(n0 + l16) * MDIM + half * 8;
  ap[1] = ap[0] + (size_t)16 * MDIM;
  bp[0] = Rh + (size_t)(b0 + l16) * MDIM + half * 8;
#pragma unroll
  for (int bi = 1; bi < 4; ++bi) bp[bi] = bp[bi - 1] + (size_t)16 * MDIM;

  for (int k0 = 0; k0 < MDIM; k0 += 32) {
    v16h a[2], bb[4];
#pragma unroll
    for (int ni = 0; ni < 2; ++ni) a[ni] = ld_frag(ap[ni] + k0, 16);
#pragma unroll
    for (int bi = 0; bi < 4; ++bi) bb[bi] = ld_frag(bp[bi] + k0, 16);
#pragma unroll
    for (int ni = 0; ni < 2; ++ni)
#pragma unroll
      for (int bi = 0; bi < 4; ++bi)
        acc[ni][bi] = wmma_f16(a[ni], bb[bi], acc[ni][bi]);
  }

  float g = gamma[it];
#pragma unroll
  for (int ni = 0; ni < 2; ++ni)
#pragma unroll
    for (int bi = 0; bi < 4; ++bi) {
      int b = b0 + bi * 16 + l16;
      int n = n0 + ni * 16 + half * 8;
      store_xpre_tile(acc[ni][bi], g, x + (size_t)b * NDIM + n,
                      Xpre + (size_t)b * NDIM + n);
    }
}

// ---------------- top-p radix select + soft-threshold + overshoot update ----------------
// one workgroup per batch column b; exact p-th largest of |xpre| via 4x 8-bit radix select

__global__ __launch_bounds__(256) void select_update_kernel(
    const float* __restrict__ Xpre,   // B x N
    float*       __restrict__ Xout,   // B x N (live x, updated in place)
    const float* __restrict__ theta, const float* __restrict__ a_arr,
    int it, int p) {
  __shared__ unsigned int av[NDIM];
  __shared__ unsigned int hist[256];
  __shared__ unsigned int s_prefix;
  __shared__ unsigned int s_want;

  const int b = blockIdx.x;
  const float* xpre = Xpre + (size_t)b * NDIM;
  float* xrow = Xout + (size_t)b * NDIM;

  for (int n = threadIdx.x; n < NDIM; n += 256)
    av[n] = __float_as_uint(fabsf(xpre[n]));
  if (threadIdx.x == 0) { s_prefix = 0u; s_want = (unsigned)p; }
  __syncthreads();

  for (int shift = 24; shift >= 0; shift -= 8) {
    hist[threadIdx.x] = 0u;
    __syncthreads();
    unsigned int prefix = s_prefix;
    for (int n = threadIdx.x; n < NDIM; n += 256) {
      unsigned int bits = av[n];
      if (shift == 24 || (bits >> (shift + 8)) == prefix)
        atomicAdd(&hist[(bits >> shift) & 255u], 1u);
    }
    __syncthreads();
    if (threadIdx.x == 0) {
      unsigned int want = s_want, cum = 0u;
      int sel = 0;
      for (int bin = 255; bin >= 0; --bin) {
        unsigned int h = hist[bin];
        if (cum + h >= want) { sel = bin; break; }
        cum += h;
      }
      s_prefix = (s_prefix << 8) | (unsigned)sel;
      s_want = want - cum;
    }
    __syncthreads();
  }

  float thresh = __uint_as_float(s_prefix);  // p-th largest |xpre| in this column
  float th = theta[it];
  float aa = a_arr[it];
  for (int n = threadIdx.x; n < NDIM; n += 256) {
    float z  = xpre[n];
    float ab = fabsf(z);
    float st = copysignf(fmaxf(ab - th, 0.0f), z);  // soft threshold
    float xn = (ab > thresh) ? z : st;              // keep top-p untouched
    float xo = xrow[n];
    float ov = 1.0f + aa / (fabsf(xn - xo) + 0.01f);
    xrow[n] = xo + ov * (xn - xo);                  // overshoot update
  }
}

// ---------------- host launcher ----------------

extern "C" void kernel_launch(void* const* d_in, const int* in_sizes, int n_in,
                              void* d_out, int out_size, void* d_ws, size_t ws_size,
                              hipStream_t stream) {
  const float* y     = (const float*)d_in[0];
  const float* phi   = (const float*)d_in[1];
  const float* W     = (const float*)d_in[2];
  const float* gamma = (const float*)d_in[3];
  const float* theta = (const float*)d_in[4];
  const float* a_arr = (const float*)d_in[5];
  const float* v_arr = (const float*)d_in[6];
  const float* vu    = (const float*)d_in[7];
  const float* th0   = (const float*)d_in[8];

  float* x = (float*)d_out;  // live x in (B, N) layout == required output x.T

  char* ws = (char*)d_ws;
  _Float16* phiH = (_Float16*)(ws);                                   //  2 MB
  _Float16* WT   = (_Float16*)(ws + (size_t)MDIM * NDIM * 2);         //  2 MB
  _Float16* U    = (_Float16*)(ws + (size_t)MDIM * NDIM * 4);         //  8 MB
  _Float16* Rh   = (_Float16*)(ws + (size_t)MDIM * NDIM * 4
                                  + (size_t)BDIM * NDIM * 2);         //  2 MB
  float*    Xpre = (float*)   (ws + (size_t)MDIM * NDIM * 4
                                  + (size_t)BDIM * NDIM * 2
                                  + (size_t)BDIM * MDIM * 2);         // 16 MB

  const int ELW4 = BDIM * NDIM / 4 / 256;           // 4096 blocks
  const int PREP = MDIM * NDIM / 256;               // 4096 blocks
  const int G1   = (MDIM / 32) * (BDIM / 64) / 4;   // 128 blocks of 4 waves
  const int G2   = (NDIM / 32) * (BDIM / 64) / 4;   // 512 blocks of 4 waves

  zero_kernel<<<ELW4, 256, 0, stream>>>(x);
  prep_kernel<<<PREP, 256, 0, stream>>>(phi, W, phiH, WT);

  for (int it = 0; it < KITER; ++it) {
    int p = 8 * (it + 1);
    if (p > NDIM) p = NDIM;
    gain_kernel<<<ELW4, 256, 0, stream>>>(x, theta, th0, v_arr, vu, it, U);
    gemm1_kernel<<<G1, 128, 0, stream>>>(phiH, U, y, Rh);
    gemm2_kernel<<<G2, 128, 0, stream>>>(WT, Rh, x, gamma, it, Xpre);
    select_update_kernel<<<BDIM, 256, 0, stream>>>(Xpre, x, theta, a_arr, it, p);
  }
}
